// MultiHeadAttentionBlock_71665824301623
// MI455X (gfx1250) — matrix operand order repaired and checked by reading, not compiled, not run on hardware
//
#include <hip/hip_runtime.h>
#include <hip/hip_bf16.h>
#include <stdint.h>

typedef __attribute__((ext_vector_type(16))) __bf16 v16bf;
typedef __attribute__((ext_vector_type(8)))  __bf16 v8bf;
typedef __attribute__((ext_vector_type(8)))  float  v8f;
typedef __attribute__((ext_vector_type(4)))  float  v4f;
typedef __attribute__((ext_vector_type(4)))  unsigned int v4u;
typedef __attribute__((ext_vector_type(8)))  int v8i;
typedef __attribute__((ext_vector_type(4)))  int v4i;

#define D_MODEL 768
#define NHEAD   12
#define DKH     64
#define SEQ     2048
#define BATCH   4
#define BS      (BATCH*SEQ)
#define LOG2E   1.44269504088896340736f

// ---------- helpers ----------

static __device__ __forceinline__ v8bf load8bf(const __bf16* p) {
    return *(const v8bf*)p;
}

static __device__ __forceinline__ v16bf cat16(v8bf a, v8bf b) {
    return __builtin_shufflevector(a, b, 0,1,2,3,4,5,6,7,8,9,10,11,12,13,14,15);
}

static __device__ __forceinline__ v8bf cvt8(const float* p) {
    v4f a = *(const v4f*)p;
    v4f b = *(const v4f*)(p + 4);
    v8bf r;
#pragma unroll
    for (int i = 0; i < 4; ++i) { r[i] = (__bf16)a[i]; r[4+i] = (__bf16)b[i]; }
    return r;
}

static __device__ __forceinline__ v8f wmma_bf16(v16bf a, v16bf b, v8f c) {
    return __builtin_amdgcn_wmma_f32_16x16x32_bf16(false, a, false, b, (short)0, c, false, false);
}

template<int MASK>
static __device__ __forceinline__ float ds_xor_f32(float v) {
    int i = __builtin_amdgcn_ds_swizzle(__builtin_bit_cast(int, v), (MASK << 10) | 0x1f);
    return __builtin_bit_cast(float, i);
}
template<int MASK>
static __device__ __forceinline__ float xor_max(float v) { return fmaxf(v, ds_xor_f32<MASK>(v)); }
template<int MASK>
static __device__ __forceinline__ float xor_add(float v) { return v + ds_xor_f32<MASK>(v); }

// ---------- TDM: async 2D tile load global->LDS (64 rows x 128B, padded to 144B rows) ----------
// D# group0: count=1, lds_addr, 57b global addr, type=2.
// D# group1: data_size=2B, pad_enable, pad_interval=32 dwords, pad_amount=4 dwords,
//            tensor_dim0=64, tensor_dim1=1<<20 (no OOB), tile=64x64, dim0_stride=768.

static __device__ __forceinline__ void tdm_load_tile64(unsigned ldsOff, const __bf16* gsrc) {
    unsigned long long ga = (unsigned long long)(uintptr_t)gsrc;
    v4u g0 = { 1u, ldsOff, (unsigned)ga,
               (unsigned)((ga >> 32) & 0x01FFFFFFu) | (2u << 30) };
    v8i g1;
    g1[0] = (int)((1u << 16) | (1u << 20) | (4u << 22) | (3u << 25));
    g1[1] = (int)(64u << 16);            // tensor_dim0 = 64
    g1[2] = 0;                           // tensor_dim1 low16 = 0
    g1[3] = (int)((1u << 4) | (64u << 16)); // tensor_dim1 = 1<<20 ; tile_dim0 = 64
    g1[4] = 64;                          // tile_dim1 = 64, tile_dim2 = 0
    g1[5] = D_MODEL;                     // tensor_dim0_stride = 768
    g1[6] = 0;
    g1[7] = 0;
    v4i gz4 = {0, 0, 0, 0};
    v8i gz8;
#pragma unroll
    for (int i = 0; i < 8; ++i) gz8[i] = 0;
    // 6-arg toolchain form: (g0, g1, g2, g3, extra, cpol)
    __builtin_amdgcn_tensor_load_to_lds(g0, g1, gz4, gz4, gz8, 0);
}

// ---------- f32 -> bf16 elementwise convert ----------

__global__ void cvt_f32_to_bf16_kernel(const float* __restrict__ src,
                                       __bf16* __restrict__ dst, int n) {
    int i = blockIdx.x * blockDim.x + threadIdx.x;
    if (i < n) dst[i] = (__bf16)src[i];
}

// ---------- GEMM: Y[m][n] = outScale * sum_k A[m][k] * W[n][k] ----------
// One wave computes a 32x64 tile: 2 A-fragments reuse each of the 4 B-fragments.

template<bool A_BF16, bool OUT_BF16>
__global__ void gemm_wmma_kernel(const void* __restrict__ Aptr,
                                 const __bf16* __restrict__ W,
                                 void* __restrict__ Yptr,
                                 int M, int N, int K, float outScale) {
    const int lane = threadIdx.x & 31;
    const int wib  = threadIdx.x >> 5;
    const int wid  = blockIdx.x * (blockDim.x >> 5) + wib;
    const int ntilesN = N / 64;
    if (wid >= (M / 32) * ntilesN) return;
    const int mtile = wid / ntilesN;
    const int ntile = wid % ntilesN;

    const int ln   = lane & 15;
    const int half = lane >> 4;
    const int mrow0 = mtile * 32 + ln;
    const int mrow1 = mrow0 + 16;

    v8f acc[2][4];
#pragma unroll
    for (int g = 0; g < 2; ++g)
#pragma unroll
        for (int c = 0; c < 4; ++c) acc[g][c] = (v8f){};

    for (int k0 = 0; k0 < K; k0 += 32) {
        v16bf a0, a1;
        if (A_BF16) {
            const __bf16* Ab0 = (const __bf16*)Aptr + (size_t)mrow0 * K + k0;
            const __bf16* Ab1 = (const __bf16*)Aptr + (size_t)mrow1 * K + k0;
            a0 = cat16(load8bf(Ab0 + half * 8), load8bf(Ab0 + 16 + half * 8));
            a1 = cat16(load8bf(Ab1 + half * 8), load8bf(Ab1 + 16 + half * 8));
        } else {
            const float* Af0 = (const float*)Aptr + (size_t)mrow0 * K + k0;
            const float* Af1 = (const float*)Aptr + (size_t)mrow1 * K + k0;
            a0 = cat16(cvt8(Af0 + half * 8), cvt8(Af0 + 16 + half * 8));
            a1 = cat16(cvt8(Af1 + half * 8), cvt8(Af1 + 16 + half * 8));
        }
        const __bf16* W0 = W + (size_t)(ntile * 64 + ln) * K + k0 + half * 8;
        v16bf b0 = cat16(load8bf(W0),                  load8bf(W0 + 16));
        v16bf b1 = cat16(load8bf(W0 + 16 * (size_t)K), load8bf(W0 + 16 * (size_t)K + 16));
        v16bf b2 = cat16(load8bf(W0 + 32 * (size_t)K), load8bf(W0 + 32 * (size_t)K + 16));
        v16bf b3 = cat16(load8bf(W0 + 48 * (size_t)K), load8bf(W0 + 48 * (size_t)K + 16));
        acc[0][0] = wmma_bf16(a0, b0, acc[0][0]);
        acc[1][0] = wmma_bf16(a1, b0, acc[1][0]);
        acc[0][1] = wmma_bf16(a0, b1, acc[0][1]);
        acc[1][1] = wmma_bf16(a1, b1, acc[1][1]);
        acc[0][2] = wmma_bf16(a0, b2, acc[0][2]);
        acc[1][2] = wmma_bf16(a1, b2, acc[1][2]);
        acc[0][3] = wmma_bf16(a0, b3, acc[0][3]);
        acc[1][3] = wmma_bf16(a1, b3, acc[1][3]);
    }

#pragma unroll
    for (int g = 0; g < 2; ++g) {
#pragma unroll
        for (int r = 0; r < 8; ++r) {
            const int row = mtile * 32 + g * 16 + r + half * 8;  // C layout: M = r + 8*half
            const size_t base = (size_t)row * N + ntile * 64 + ln;
            float y0 = acc[g][0][r] * outScale, y1 = acc[g][1][r] * outScale;
            float y2 = acc[g][2][r] * outScale, y3 = acc[g][3][r] * outScale;
            if (OUT_BF16) {
                __bf16* Y = (__bf16*)Yptr;
                Y[base] = (__bf16)y0; Y[base+16] = (__bf16)y1;
                Y[base+32] = (__bf16)y2; Y[base+48] = (__bf16)y3;
            } else {
                float* Y = (float*)Yptr;
                Y[base] = y0; Y[base+16] = y1; Y[base+32] = y2; Y[base+48] = y3;
            }
        }
    }
}

// ---------- Flash attention (bf16 Q/K/V in [BS,768] layout, head = 64 cols) ----------
// Block = 128 threads (4 waves) = 64 query rows of one (b,h).
// K/V streamed in 64-key chunks via TDM (double buffered): wave0 issues K D#,
// wave1 issues V D#; each waits its own TENSORcnt before the workgroup barrier.
// Scores already carry the log2(e)/sqrt(dk) scale -> exp2-domain softmax.

__global__ __launch_bounds__(128)
void attn_flash_kernel(const __bf16* __restrict__ Qg,
                       const __bf16* __restrict__ Kg,
                       const __bf16* __restrict__ Vg,
                       __bf16* __restrict__ Ctx) {
    const int qblocks = SEQ / 64;                       // 32
    int bid = blockIdx.x;
    const int qb = bid % qblocks;  bid /= qblocks;
    const int h  = bid % NHEAD;    bid /= NHEAD;
    const int b  = bid;

    const int lane = threadIdx.x & 31;
    const int wave = threadIdx.x >> 5;                  // 0..3
    const int ln   = lane & 15;
    const int half = lane >> 4;

    // padded rows: 72 elems = 144B (16B aligned, 36-bank stride -> conflict free)
    __shared__ __align__(16) __bf16 Kl[2][64][72];      // K chunk, row-major [key][dk]
    __shared__ __align__(16) __bf16 Vs[2][64][72];      // V chunk, row-major [key][dk]
    __shared__ __align__(16) __bf16 Vt[64][72];         // V chunk transposed [dk][key]
    __shared__ __align__(16) __bf16 Pl[4][16][72];      // per-wave P tile [query][key]

    const size_t hb = (size_t)b * SEQ * D_MODEL + (size_t)h * DKH;
    const __bf16* Qb = Qg + hb;
    const __bf16* Kb = Kg + hb;
    const __bf16* Vb = Vg + hb;

    // Q fragments (16x64 = two 16x32 A-fragments), resident in VGPRs
    const int qrow = qb * 64 + wave * 16 + ln;
    const __bf16* Qr = Qb + (size_t)qrow * D_MODEL;
    const v16bf qa0 = cat16(load8bf(Qr +      half * 8), load8bf(Qr + 16 + half * 8));
    const v16bf qa1 = cat16(load8bf(Qr + 32 + half * 8), load8bf(Qr + 48 + half * 8));

    v8f o0 = {}, o1 = {}, o2 = {}, o3 = {};
    float mrow[8], lrow[8];
#pragma unroll
    for (int r = 0; r < 8; ++r) { mrow[r] = -3.0e38f; lrow[r] = 0.0f; }

    const unsigned klOff[2] = { (unsigned)(uintptr_t)&Kl[0][0][0],
                                (unsigned)(uintptr_t)&Kl[1][0][0] };
    const unsigned vsOff[2] = { (unsigned)(uintptr_t)&Vs[0][0][0],
                                (unsigned)(uintptr_t)&Vs[1][0][0] };

    // prologue: stage chunk 0
    if (wave == 0)      tdm_load_tile64(klOff[0], Kb);
    else if (wave == 1) tdm_load_tile64(vsOff[0], Vb);

    int buf = 0;
    for (int kc = 0; kc < SEQ; kc += 64, buf ^= 1) {
        if (wave < 2) __builtin_amdgcn_s_wait_tensorcnt(0);
        __syncthreads();                                // (A) chunk `buf` resident

        if (kc + 64 < SEQ) {                            // stage next chunk into other buffer
            if (wave == 0)      tdm_load_tile64(klOff[buf ^ 1], Kb + (size_t)(kc + 64) * D_MODEL);
            else if (wave == 1) tdm_load_tile64(vsOff[buf ^ 1], Vb + (size_t)(kc + 64) * D_MODEL);
        }

        // ---- cooperative transpose Vs[buf] -> Vt ----
        {
            const int key  = threadIdx.x >> 1;          // 0..63
            const int hseg = (threadIdx.x & 1) * 32;    // dk half
            const __bf16* src = &Vs[buf][key][hseg];
            v8bf t0 = load8bf(src), t1 = load8bf(src + 8);
            v8bf t2 = load8bf(src + 16), t3 = load8bf(src + 24);
#pragma unroll
            for (int j = 0; j < 8; ++j) {
                Vt[hseg +      j][key] = t0[j];
                Vt[hseg +  8 + j][key] = t1[j];
                Vt[hseg + 16 + j][key] = t2[j];
                Vt[hseg + 24 + j][key] = t3[j];
            }
        }
        __syncthreads();                                // (B) Vt ready

        // ---- scores S(16x64) = Q(16x64) . K_chunk^T, B-frags from LDS ----
        v8f s0 = {}, s1 = {}, s2 = {}, s3 = {};
#define KFRAG(T, KS) cat16(load8bf(&Kl[buf][(T)*16 + ln][(KS)*32 + half*8]), \
                           load8bf(&Kl[buf][(T)*16 + ln][(KS)*32 + half*8 + 16]))
        s0 = wmma_bf16(qa0, KFRAG(0,0), s0);  s0 = wmma_bf16(qa1, KFRAG(0,1), s0);
        s1 = wmma_bf16(qa0, KFRAG(1,0), s1);  s1 = wmma_bf16(qa1, KFRAG(1,1), s1);
        s2 = wmma_bf16(qa0, KFRAG(2,0), s2);  s2 = wmma_bf16(qa1, KFRAG(2,1), s2);
        s3 = wmma_bf16(qa0, KFRAG(3,0), s3);  s3 = wmma_bf16(qa1, KFRAG(3,1), s3);
#undef KFRAG

        // ---- online softmax in exp2 domain ----
#pragma unroll
        for (int r = 0; r < 8; ++r) {
            float cm = fmaxf(fmaxf(s0[r], s1[r]), fmaxf(s2[r], s3[r]));
            cm = xor_max<1>(cm); cm = xor_max<2>(cm);
            cm = xor_max<4>(cm); cm = xor_max<8>(cm);
            const float mnew  = fmaxf(mrow[r], cm);
            const float scale = exp2f(mrow[r] - mnew);
            const float p0 = exp2f(s0[r] - mnew);
            const float p1 = exp2f(s1[r] - mnew);
            const float p2 = exp2f(s2[r] - mnew);
            const float p3 = exp2f(s3[r] - mnew);
            float ps = (p0 + p1) + (p2 + p3);
            ps = xor_add<1>(ps); ps = xor_add<2>(ps);
            ps = xor_add<4>(ps); ps = xor_add<8>(ps);
            lrow[r] = lrow[r] * scale + ps;
            mrow[r] = mnew;
            o0[r] *= scale; o1[r] *= scale; o2[r] *= scale; o3[r] *= scale;
            const int m = r + half * 8;                 // C layout row
            Pl[wave][m][ln]      = (__bf16)p0;
            Pl[wave][m][16 + ln] = (__bf16)p1;
            Pl[wave][m][32 + ln] = (__bf16)p2;
            Pl[wave][m][48 + ln] = (__bf16)p3;
        }
        asm volatile("s_wait_dscnt 0" ::: "memory");    // order wave-local P store -> load

        // ---- O += P(16x64) . V_chunk(64x64) ----
        const v16bf pf0 = cat16(load8bf(&Pl[wave][ln][     half * 8]),
                                load8bf(&Pl[wave][ln][16 + half * 8]));
        const v16bf pf1 = cat16(load8bf(&Pl[wave][ln][32 + half * 8]),
                                load8bf(&Pl[wave][ln][48 + half * 8]));
#define VFRAG(T, KS) cat16(load8bf(&Vt[(T)*16 + ln][(KS)*32 + half*8]), \
                           load8bf(&Vt[(T)*16 + ln][(KS)*32 + half*8 + 16]))
        o0 = wmma_bf16(pf0, VFRAG(0,0), o0);  o0 = wmma_bf16(pf1, VFRAG(0,1), o0);
        o1 = wmma_bf16(pf0, VFRAG(1,0), o1);  o1 = wmma_bf16(pf1, VFRAG(1,1), o1);
        o2 = wmma_bf16(pf0, VFRAG(2,0), o2);  o2 = wmma_bf16(pf1, VFRAG(2,1), o2);
        o3 = wmma_bf16(pf0, VFRAG(3,0), o3);  o3 = wmma_bf16(pf1, VFRAG(3,1), o3);
#undef VFRAG
    }

    // ---- epilogue: normalize and write context (bf16) ----
#pragma unroll
    for (int r = 0; r < 8; ++r) {
        const float inv = 1.0f / lrow[r];
        const int m = qb * 64 + wave * 16 + r + half * 8;
        const size_t base = ((size_t)b * SEQ + m) * D_MODEL + (size_t)h * DKH + ln;
        Ctx[base]      = (__bf16)(o0[r] * inv);
        Ctx[base + 16] = (__bf16)(o1[r] * inv);
        Ctx[base + 32] = (__bf16)(o2[r] * inv);
        Ctx[base + 48] = (__bf16)(o3[r] * inv);
    }
}

// ---------- host launcher ----------

extern "C" void kernel_launch(void* const* d_in, const int* in_sizes, int n_in,
                              void* d_out, int out_size, void* d_ws, size_t ws_size,
                              hipStream_t stream) {
    const float* q  = (const float*)d_in[0];
    const float* k  = (const float*)d_in[1];
    const float* v  = (const float*)d_in[2];
    const float* wq = (const float*)d_in[3];
    const float* wk = (const float*)d_in[4];
    const float* wv = (const float*)d_in[5];
    const float* wo = (const float*)d_in[6];
    float* out = (float*)d_out;

    char* ws = (char*)d_ws;
    const size_t matB   = (size_t)BS * D_MODEL * sizeof(__bf16);      // 12.6 MB
    const size_t wElems = (size_t)D_MODEL * D_MODEL;
    __bf16* wsQ  = (__bf16*)(ws);
    __bf16* wsK  = (__bf16*)(ws + 1 * matB);
    __bf16* wsV  = (__bf16*)(ws + 2 * matB);
    __bf16* wsC  = (__bf16*)(ws + 3 * matB);
    __bf16* wsWq = (__bf16*)(ws + 4 * matB);
    __bf16* wsWk = wsWq + wElems;
    __bf16* wsWv = wsWk + wElems;
    __bf16* wsWo = wsWv + wElems;

    // 1) weights -> bf16
    const int cvtBlocks = (int)((wElems + 255) / 256);
    cvt_f32_to_bf16_kernel<<<cvtBlocks, 256, 0, stream>>>(wq, wsWq, (int)wElems);
    cvt_f32_to_bf16_kernel<<<cvtBlocks, 256, 0, stream>>>(wk, wsWk, (int)wElems);
    cvt_f32_to_bf16_kernel<<<cvtBlocks, 256, 0, stream>>>(wv, wsWv, (int)wElems);
    cvt_f32_to_bf16_kernel<<<cvtBlocks, 256, 0, stream>>>(wo, wsWo, (int)wElems);

    // 2) projections (f32 A, bf16 out); fold 1/sqrt(d_k) * log2(e) into Q for exp2 softmax
    const int totalTiles = (BS / 32) * (D_MODEL / 64);   // 3072 waves
    const int gemmBlocks = totalTiles / 8;               // 384 blocks x 256 thr
    gemm_wmma_kernel<false, true><<<gemmBlocks, 256, 0, stream>>>(q, wsWq, wsQ, BS, D_MODEL, D_MODEL, 0.125f * LOG2E);
    gemm_wmma_kernel<false, true><<<gemmBlocks, 256, 0, stream>>>(k, wsWk, wsK, BS, D_MODEL, D_MODEL, 1.0f);
    gemm_wmma_kernel<false, true><<<gemmBlocks, 256, 0, stream>>>(v, wsWv, wsV, BS, D_MODEL, D_MODEL, 1.0f);

    // 3) attention (TDM-staged K/V, double buffered)
    attn_flash_kernel<<<BATCH * NHEAD * (SEQ / 64), 128, 0, stream>>>(wsQ, wsK, wsV, wsC);

    // 4) output projection (bf16 A, f32 out)
    gemm_wmma_kernel<true, false><<<gemmBlocks, 256, 0, stream>>>(wsC, wsWo, out, BS, D_MODEL, D_MODEL, 1.0f);
}
